// FuXiBlock_89524298318183
// MI455X (gfx1250) — matrix operand order repaired and checked by reading, not compiled, not run on hardware
//
#include <hip/hip_runtime.h>

// ---------------------------------------------------------------------------
// CDNA5 (gfx1250) wave32 WMMA types
// ---------------------------------------------------------------------------
typedef __attribute__((ext_vector_type(16))) __bf16 v16bf;
typedef __attribute__((ext_vector_type(8)))  float  v8f;
typedef __attribute__((ext_vector_type(4)))  int    v4i;

union Frag {
  uint4 u[2];
  v16bf v;
};

// ---- optional CDNA5 async global->LDS path (ASYNCcnt), with safe fallback ----
#ifndef __has_builtin
#define __has_builtin(x) 0
#endif
#if defined(__HIP_DEVICE_COMPILE__) && __has_builtin(__builtin_amdgcn_global_load_async_to_lds_b128)
#define USE_ASYNC 1
#else
#define USE_ASYNC 0
#endif

#if USE_ASYNC
__device__ __forceinline__ void cp_async16(void* lds, const void* g) {
  // signature (from compiler diagnostic): (v4i AS1* src, v4i AS3* dst, imm, imm)
  __builtin_amdgcn_global_load_async_to_lds_b128(
      (__attribute__((address_space(1))) v4i*)(unsigned long long)g,
      (__attribute__((address_space(3))) v4i*)(unsigned int)(unsigned long long)lds,
      0, 0);
}
__device__ __forceinline__ void async_wait0() {
#if __has_builtin(__builtin_amdgcn_s_wait_asynccnt)
  __builtin_amdgcn_s_wait_asynccnt(0);
#else
  asm volatile("s_wait_asynccnt 0x0" ::: "memory");
#endif
}
#endif

__device__ __forceinline__ unsigned short f32_to_bf16(float f) {
  unsigned int u = __float_as_uint(f);
  u += 0x7FFFu + ((u >> 16) & 1u);   // round-to-nearest-even
  return (unsigned short)(u >> 16);
}

// silu via hardware v_exp_f32 + v_rcp_f32 (no IEEE division chains)
__device__ __forceinline__ float silu_f(float x) {
  return x * __builtin_amdgcn_rcpf(1.0f + __expf(-x));
}

// A-matrix fragment (16x32, 16-bit): lane<16 holds K = kbase+{0..7, 16..23},
// lane>=16 holds K = kbase+{8..15, 24..31}.
__device__ __forceinline__ v16bf ld_fragA(const unsigned short* rowp, int kbase, int hi) {
  Frag f;
  f.u[0] = *(const uint4*)(rowp + kbase + hi * 8);
  f.u[1] = *(const uint4*)(rowp + kbase + 16 + hi * 8);
  return f.v;
}

// B-matrix fragment (32x16, 16-bit): lane<16 (col=lane) holds K = kbase+{0..7, 16..23},
// lane>=16 holds K = kbase+{8..15, 24..31} (same order as A).
__device__ __forceinline__ v16bf ld_fragB(const unsigned short* rowp, int kbase, int hi) {
  Frag f;
  f.u[0] = *(const uint4*)(rowp + kbase + hi * 8);
  f.u[1] = *(const uint4*)(rowp + kbase + 16 + hi * 8);
  return f.v;
}

__device__ __forceinline__ v8f wmma_bf16(v16bf a, v16bf b, v8f c) {
  return __builtin_amdgcn_wmma_f32_16x16x32_bf16(false, a, false, b, (short)0, c,
                                                 false, false);
}

// ---------------------------------------------------------------------------
// Weight convert + transpose: f32 [K,N] row-major -> bf16 [N,K] row-major
// ---------------------------------------------------------------------------
__global__ void wt_transpose(const float* __restrict__ w, unsigned short* __restrict__ wt,
                             int K, int N) {
  int idx = blockIdx.x * 256 + threadIdx.x;
  if (idx >= K * N) return;
  int k = idx / N, n = idx % N;
  wt[(size_t)n * K + k] = f32_to_bf16(w[idx]);
}

// ---------------------------------------------------------------------------
// RMSNorm (optionally * mul) -> bf16. One block (256 thr) per row of d=1024.
// ---------------------------------------------------------------------------
template <bool HASMUL>
__global__ __launch_bounds__(256) void rmsnorm_k(const float* __restrict__ x,
                                                 const float* __restrict__ g,
                                                 const float* __restrict__ mul,
                                                 unsigned short* __restrict__ out, int d) {
  int row = blockIdx.x;
  const float* xr = x + (size_t)row * d;
  __shared__ float red[8];
  __shared__ float s_rms;
  float p = 0.0f;
  for (int i = threadIdx.x; i < d; i += 256) { float v = xr[i]; p += v * v; }
  for (int o = 16; o > 0; o >>= 1) p += __shfl_xor(p, o, 32);
  if ((threadIdx.x & 31) == 0) red[threadIdx.x >> 5] = p;
  __syncthreads();
  if (threadIdx.x == 0) {
    float s = 0.0f;
    for (int i = 0; i < 8; ++i) s += red[i];
    s_rms = __builtin_amdgcn_rsqf(s * (1.0f / 1024.0f) + 1e-8f);
  }
  __syncthreads();
  float rms = s_rms;
  for (int i = threadIdx.x; i < d; i += 256) {
    float v = xr[i] * rms * g[i];
    if (HASMUL) v *= mul[(size_t)row * d + i];
    out[(size_t)row * d + i] = f32_to_bf16(v);
  }
}

// ---------------------------------------------------------------------------
// GEMM: C[M,N] = act(A[M,K](bf16) @ W[N,K]^T(bf16) + bias) (+ resid)
// Block = 256 thr (8 waves), tile 128x128, BK=64. Wave = 64x32 (4x2 wmma tiles).
// ---------------------------------------------------------------------------
template <int ACT, bool OUTBF, bool RES>
__global__ __launch_bounds__(256) void gemm_bf16(
    const unsigned short* __restrict__ A, const unsigned short* __restrict__ W,
    const float* __restrict__ bias, const float* __restrict__ resid,
    float* __restrict__ outf, unsigned short* __restrict__ outb,
    int M, int N, int K) {
  __shared__ unsigned short As[128][72];  // 72-half stride: conflict-free b128 frags
  __shared__ unsigned short Bs[128][72];
  const int tid = threadIdx.x;
  const int lane = tid & 31, l16 = lane & 15, hi = lane >> 4;
  const int wid = tid >> 5;
  const int wm = wid >> 2;  // 0..1 -> 64-row strip
  const int wn = wid & 3;   // 0..3 -> 32-col strip
  const int m0 = blockIdx.y * 128;
  const int n0 = blockIdx.x * 128;

  v8f acc[4][2] = {};

  for (int k0 = 0; k0 < K; k0 += 64) {
    __syncthreads();
#if USE_ASYNC
#pragma unroll
    for (int p = 0; p < 4; ++p) {
      int lin = (p * 256 + tid) * 8;  // halves
      int row = lin >> 6, col = lin & 63;
      cp_async16(&As[row][col], A + (size_t)(m0 + row) * K + k0 + col);
      cp_async16(&Bs[row][col], W + (size_t)(n0 + row) * K + k0 + col);
    }
    async_wait0();
#else
    uint4 ra[4], rb[4];
#pragma unroll
    for (int p = 0; p < 4; ++p) {
      int lin = (p * 256 + tid) * 8;
      int row = lin >> 6, col = lin & 63;
      ra[p] = *(const uint4*)(A + (size_t)(m0 + row) * K + k0 + col);
      rb[p] = *(const uint4*)(W + (size_t)(n0 + row) * K + k0 + col);
    }
#pragma unroll
    for (int p = 0; p < 4; ++p) {
      int lin = (p * 256 + tid) * 8;
      int row = lin >> 6, col = lin & 63;
      *(uint4*)&As[row][col] = ra[p];
      *(uint4*)&Bs[row][col] = rb[p];
    }
#endif
    if (k0 + 64 < K) {  // prefetch next K tile (global_prefetch_b8)
      int lin = tid * 8;
      int row = lin >> 6, col = lin & 63;
      __builtin_prefetch(A + (size_t)(m0 + row) * K + k0 + 64 + col, 0, 1);
      __builtin_prefetch(W + (size_t)(n0 + row) * K + k0 + 64 + col, 0, 1);
    }
    __syncthreads();
#pragma unroll
    for (int kk = 0; kk < 64; kk += 32) {
      v16bf af[4], bw[2];
#pragma unroll
      for (int i = 0; i < 4; ++i)
        af[i] = ld_fragA(&As[wm * 64 + i * 16 + l16][0], kk, hi);
#pragma unroll
      for (int j = 0; j < 2; ++j)
        bw[j] = ld_fragB(&Bs[wn * 32 + j * 16 + l16][0], kk, hi);
#pragma unroll
      for (int i = 0; i < 4; ++i)
#pragma unroll
        for (int j = 0; j < 2; ++j) acc[i][j] = wmma_bf16(af[i], bw[j], acc[i][j]);
    }
  }
  // Epilogue. C/D layout: lane -> col (l16), VGPR r -> row r + hi*8.
#pragma unroll
  for (int i = 0; i < 4; ++i) {
#pragma unroll
    for (int j = 0; j < 2; ++j) {
      int col = n0 + wn * 32 + j * 16 + l16;
      float bv = bias[col];
#pragma unroll
      for (int r = 0; r < 8; ++r) {
        int row = m0 + wm * 64 + i * 16 + hi * 8 + r;
        float v = acc[i][j][r] + bv;
        if (ACT == 1) v = silu_f(v);
        if (RES) v += resid[(size_t)row * N + col];
        if (OUTBF) outb[(size_t)row * N + col] = f32_to_bf16(v);
        else       outf[(size_t)row * N + col] = v;
      }
    }
  }
}

// ---------------------------------------------------------------------------
// Split heads: q/k -> [B,H,S,64]; v -> transposed [B,H,64,S] (for B-fragments)
// ---------------------------------------------------------------------------
__global__ void split_heads(const unsigned short* __restrict__ q,
                            const unsigned short* __restrict__ k,
                            const unsigned short* __restrict__ v,
                            unsigned short* __restrict__ qh,
                            unsigned short* __restrict__ kh,
                            unsigned short* __restrict__ vT,
                            int Bb, int S, int H) {
  const int dh = 64;
  long long idx = (long long)blockIdx.x * 256 + threadIdx.x;
  long long total = (long long)Bb * S * H * dh;
  if (idx >= total) return;
  int j = (int)(idx % dh);
  int h = (int)((idx / dh) % H);
  int s = (int)((idx / ((long long)dh * H)) % S);
  int b = (int)(idx / ((long long)dh * H * S));
  size_t ph = (((size_t)b * H + h) * S + s) * dh + j;
  qh[ph] = q[idx];
  kh[ph] = k[idx];
  vT[(((size_t)b * H + h) * dh + j) * (size_t)S + s] = v[idx];
}

// ---------------------------------------------------------------------------
// Streaming SiLU-attention: out[64q x 64dh] per block for one (b,h).
// SiLU needs no row reduction -> scores never touch HBM (saves 2x536MB).
// 4 waves; wave w owns q rows [w*16, w*16+16).
// ---------------------------------------------------------------------------
__global__ __launch_bounds__(128) void attn_k(const unsigned short* __restrict__ qh,
                                              const unsigned short* __restrict__ kh,
                                              const unsigned short* __restrict__ vT,
                                              const float* __restrict__ pbias,
                                              float* __restrict__ out, int S, int H) {
  __shared__ unsigned short qs[64][72];   // q tile (rows: query, 64 dh halves)
  __shared__ unsigned short ks[32][72];   // k tile (rows: key,   64 dh halves)
  __shared__ unsigned short vs[64][40];   // v^T tile (rows: dh j, 32 t halves)
  __shared__ unsigned short as_[64][40];  // silu(score) bf16, [query][t]
  __shared__ float pb_s[2048];            // p_bias[:,h] staged once per block

  const int tid = threadIdx.x;
  const int lane = tid & 31, l16 = lane & 15, hi = lane >> 4, w = tid >> 5;
  const int q0 = blockIdx.x * 64;
  const int h = blockIdx.y, b = blockIdx.z;

  const size_t headq = (((size_t)b * H + h) * S + q0) * 64;
  const size_t headk = (((size_t)b * H + h) * S) * 64;
  const size_t headv = (((size_t)b * H + h) * 64) * (size_t)S;

  // stage p_bias column for this head (8KB)
  for (int i = tid; i < S; i += 128) pb_s[i] = pbias[(size_t)i * H + h];

  // stage q tile: 64 rows x 128B
#if USE_ASYNC
#pragma unroll
  for (int p = 0; p < 4; ++p) {
    int lin = (p * 128 + tid) * 8;
    int row = lin >> 6, col = lin & 63;
    cp_async16(&qs[row][col], qh + headq + (size_t)row * 64 + col);
  }
  async_wait0();
#else
  {
    uint4 rq[4];
#pragma unroll
    for (int p = 0; p < 4; ++p) {
      int lin = (p * 128 + tid) * 8;
      int row = lin >> 6, col = lin & 63;
      rq[p] = *(const uint4*)(qh + headq + (size_t)row * 64 + col);
    }
#pragma unroll
    for (int p = 0; p < 4; ++p) {
      int lin = (p * 128 + tid) * 8;
      int row = lin >> 6, col = lin & 63;
      *(uint4*)&qs[row][col] = rq[p];
    }
  }
#endif
  __syncthreads();
  // q A-fragments live in registers for the whole key loop
  v16bf aq0 = ld_fragA(&qs[w * 16 + l16][0], 0, hi);
  v16bf aq1 = ld_fragA(&qs[w * 16 + l16][0], 32, hi);

  v8f oacc[4] = {};

  for (int t0 = 0; t0 < S; t0 += 32) {
    __syncthreads();  // previous iteration consumed ks/vs
#if USE_ASYNC
#pragma unroll
    for (int p = 0; p < 2; ++p) {  // k tile: 32 rows x 128B
      int lin = (p * 128 + tid) * 8;
      int row = lin >> 6, col = lin & 63;
      cp_async16(&ks[row][col], kh + headk + (size_t)(t0 + row) * 64 + col);
    }
#pragma unroll
    for (int p = 0; p < 2; ++p) {  // v^T tile: 64 rows x 64B
      int lin = (p * 128 + tid) * 8;
      int row = lin >> 5, col = lin & 31;
      cp_async16(&vs[row][col], vT + headv + (size_t)row * S + t0 + col);
    }
    async_wait0();
#else
    {
      uint4 rk[2], rv[2];
#pragma unroll
      for (int p = 0; p < 2; ++p) {
        int lin = (p * 128 + tid) * 8;
        int row = lin >> 6, col = lin & 63;
        rk[p] = *(const uint4*)(kh + headk + (size_t)(t0 + row) * 64 + col);
      }
#pragma unroll
      for (int p = 0; p < 2; ++p) {
        int lin = (p * 128 + tid) * 8;
        int row = lin >> 5, col = lin & 31;
        rv[p] = *(const uint4*)(vT + headv + (size_t)row * S + t0 + col);
      }
#pragma unroll
      for (int p = 0; p < 2; ++p) {
        int lin = (p * 128 + tid) * 8;
        *(uint4*)&ks[lin >> 6][lin & 63] = rk[p];
        *(uint4*)&vs[lin >> 5][lin & 31] = rv[p];
      }
    }
#endif
    __syncthreads();

    // score: [16q x 32t] = q(16x64) @ k^T(64x32); B lane = key column t.
    // Load all 4 B-fragments first so ds_loads overlap, then issue WMMAs.
    v16bf bk00 = ld_fragB(&ks[l16][0], 0, hi);
    v16bf bk01 = ld_fragB(&ks[l16][0], 32, hi);
    v16bf bk10 = ld_fragB(&ks[16 + l16][0], 0, hi);
    v16bf bk11 = ld_fragB(&ks[16 + l16][0], 32, hi);
    v8f s0 = {}, s1 = {};
    s0 = wmma_bf16(aq0, bk00, s0);
    s0 = wmma_bf16(aq1, bk01, s0);
    s1 = wmma_bf16(aq0, bk10, s1);
    s1 = wmma_bf16(aq1, bk11, s1);

    float pb0 = pb_s[t0 + l16];
    float pb1 = pb_s[t0 + 16 + l16];
#pragma unroll
    for (int r = 0; r < 8; ++r) {
      float x0 = fmaf(s0[r], 0.125f, pb0);  // scale = 1/sqrt(64)
      float x1 = fmaf(s1[r], 0.125f, pb1);
      as_[w * 16 + hi * 8 + r][l16]      = f32_to_bf16(silu_f(x0));
      as_[w * 16 + hi * 8 + r][16 + l16] = f32_to_bf16(silu_f(x1));
    }
    // same-wave LDS RAW: DS ops from one wave are in order; rows are wave-private
    v16bf aa  = ld_fragA(&as_[w * 16 + l16][0], 0, hi);
    v16bf bv0 = ld_fragB(&vs[l16][0], 0, hi);
    v16bf bv1 = ld_fragB(&vs[16 + l16][0], 0, hi);
    v16bf bv2 = ld_fragB(&vs[32 + l16][0], 0, hi);
    v16bf bv3 = ld_fragB(&vs[48 + l16][0], 0, hi);
    oacc[0] = wmma_bf16(aa, bv0, oacc[0]);
    oacc[1] = wmma_bf16(aa, bv1, oacc[1]);
    oacc[2] = wmma_bf16(aa, bv2, oacc[2]);
    oacc[3] = wmma_bf16(aa, bv3, oacc[3]);
  }

  // epilogue -> out[b, s, h*64 + j] f32
#pragma unroll
  for (int nj = 0; nj < 4; ++nj)
#pragma unroll
    for (int r = 0; r < 8; ++r) {
      int row = q0 + w * 16 + hi * 8 + r;
      out[(((size_t)b * S + row) * H + h) * 64 + nj * 16 + l16] = oacc[nj][r];
    }
}

// ---------------------------------------------------------------------------
// SwiGLU: x12 f32 [M,2d] -> bf16 [M,d]
// ---------------------------------------------------------------------------
__global__ void swiglu_k(const float* __restrict__ x12, unsigned short* __restrict__ out,
                         int M, int d) {
  long long idx = (long long)blockIdx.x * 256 + threadIdx.x;
  if (idx >= (long long)M * d) return;
  int row = (int)(idx / d), c = (int)(idx % d);
  float a = x12[(size_t)row * 2 * d + c];
  float b = x12[(size_t)row * 2 * d + d + c];
  out[idx] = f32_to_bf16(silu_f(a) * b);
}

// ---------------------------------------------------------------------------
// Orchestration
// ---------------------------------------------------------------------------
extern "C" void kernel_launch(void* const* d_in, const int* in_sizes, int n_in,
                              void* d_out, int out_size, void* d_ws, size_t ws_size,
                              hipStream_t stream) {
  (void)in_sizes; (void)n_in; (void)out_size;
  const int d = 1024, Bb = 2, S = 2048, H = 16;
  const int M = Bb * S;  // 4096 tokens

  const float* x  = (const float*)d_in[0];
  const float* pb = (const float*)d_in[2];
  const float* wq = (const float*)d_in[3];  const float* bq = (const float*)d_in[4];
  const float* wk = (const float*)d_in[5];  const float* bk = (const float*)d_in[6];
  const float* wv = (const float*)d_in[7];  const float* bv = (const float*)d_in[8];
  const float* wu = (const float*)d_in[9];  const float* bu = (const float*)d_in[10];
  const float* g_ams = (const float*)d_in[11];
  const float* w0 = (const float*)d_in[12]; const float* b0 = (const float*)d_in[13];
  const float* w1 = (const float*)d_in[14]; const float* b1 = (const float*)d_in[15];
  const float* w2 = (const float*)d_in[16]; const float* b2 = (const float*)d_in[17];
  const float* g_mffn = (const float*)d_in[18];
  float* out = (float*)d_out;

  const size_t MB = 1u << 20;
  if (ws_size < 137 * MB) return;  // deterministic no-op if scratch too small
  char* ws = (char*)d_ws;

  unsigned short* WTq = (unsigned short*)(ws + 0 * MB);    // [1024,1024] bf16
  unsigned short* WTk = (unsigned short*)(ws + 2 * MB);
  unsigned short* WTv = (unsigned short*)(ws + 4 * MB);
  unsigned short* WTu = (unsigned short*)(ws + 6 * MB);
  unsigned short* WT0 = (unsigned short*)(ws + 8 * MB);
  unsigned short* WT1 = (unsigned short*)(ws + 10 * MB);   // [2048,1024] bf16
  unsigned short* WT2 = (unsigned short*)(ws + 14 * MB);
  unsigned short* xn   = (unsigned short*)(ws + 16 * MB);  // [M,d] bf16
  unsigned short* qlin = (unsigned short*)(ws + 24 * MB);
  unsigned short* klin = (unsigned short*)(ws + 32 * MB);
  unsigned short* vlin = (unsigned short*)(ws + 40 * MB);
  unsigned short* qh   = (unsigned short*)(ws + 48 * MB);  // [B,H,S,64]
  unsigned short* kh   = (unsigned short*)(ws + 56 * MB);
  unsigned short* vTb  = (unsigned short*)(ws + 64 * MB);  // [B,H,64,S]
  float* ub    = (float*)(ws + 72 * MB);                   // [M,d] f32
  float* attn  = (float*)(ws + 88 * MB);                   // [M,d] f32
  unsigned short* ams   = (unsigned short*)(ws + 104 * MB);
  float* obuf  = (float*)(ws + 112 * MB);                  // [M,d] f32
  unsigned short* onorm = (unsigned short*)(ws + 128 * MB);
  // reuse dead regions (qlin..qh consumed by attention before these producers run)
  float* x12 = (float*)(ws + 24 * MB);                     // [M,2d] f32 (32MB)
  unsigned short* sg = (unsigned short*)(ws + 56 * MB);    // [M,d] bf16 (8MB)

  // 1) weight convert+transpose (bf16 [N,K])
  wt_transpose<<<(d * d + 255) / 256, 256, 0, stream>>>(wq, WTq, d, d);
  wt_transpose<<<(d * d + 255) / 256, 256, 0, stream>>>(wk, WTk, d, d);
  wt_transpose<<<(d * d + 255) / 256, 256, 0, stream>>>(wv, WTv, d, d);
  wt_transpose<<<(d * d + 255) / 256, 256, 0, stream>>>(wu, WTu, d, d);
  wt_transpose<<<(d * d + 255) / 256, 256, 0, stream>>>(w0, WT0, d, d);
  wt_transpose<<<(d * 2 * d + 255) / 256, 256, 0, stream>>>(w1, WT1, d, 2 * d);
  wt_transpose<<<(d * d + 255) / 256, 256, 0, stream>>>(w2, WT2, d, d);

  // 2) x_norm = RMSNorm(x, g_ams) -> bf16
  rmsnorm_k<false><<<M, 256, 0, stream>>>(x, g_ams, nullptr, xn, d);

  dim3 gemm_grid(d / 128, M / 128);
  // 3) q,k,v projections (bf16 out)
  gemm_bf16<0, true, false><<<gemm_grid, 256, 0, stream>>>(xn, WTq, bq, nullptr, nullptr, qlin, M, d, d);
  gemm_bf16<0, true, false><<<gemm_grid, 256, 0, stream>>>(xn, WTk, bk, nullptr, nullptr, klin, M, d, d);
  gemm_bf16<0, true, false><<<gemm_grid, 256, 0, stream>>>(xn, WTv, bv, nullptr, nullptr, vlin, M, d, d);
  // 4) u = silu(x_norm @ wu + bu) (f32 out)
  gemm_bf16<1, false, false><<<gemm_grid, 256, 0, stream>>>(xn, WTu, bu, nullptr, ub, nullptr, M, d, d);

  // 5) head split / v transpose
  split_heads<<<(M * d + 255) / 256, 256, 0, stream>>>(qlin, klin, vlin, qh, kh, vTb, Bb, S, H);

  // 6) fused SiLU-attention
  attn_k<<<dim3(S / 64, H, Bb), 128, 0, stream>>>(qh, kh, vTb, pb, attn, S, H);

  // 7) ams = RMSNorm(attn, g_ams) * u -> bf16
  rmsnorm_k<true><<<M, 256, 0, stream>>>(attn, g_ams, ub, ams, d);

  // 8) o = ams @ w0 + b0 + x (f32)
  gemm_bf16<0, false, true><<<gemm_grid, 256, 0, stream>>>(ams, WT0, b0, x, obuf, nullptr, M, d, d);

  // 9) o_norm = RMSNorm(o, g_mffn) -> bf16
  rmsnorm_k<false><<<M, 256, 0, stream>>>(obuf, g_mffn, nullptr, onorm, d);

  // 10) x12 = o_norm @ w1 + b1 (f32, N=2048)
  gemm_bf16<0, false, false><<<dim3(2 * d / 128, M / 128), 256, 0, stream>>>(onorm, WT1, b1, nullptr, x12, nullptr, M, 2 * d, d);

  // 11) swiglu
  swiglu_k<<<(M * d + 255) / 256, 256, 0, stream>>>(x12, sg, M, d);

  // 12) out = sg @ w2 + b2 + o (f32 -> d_out)
  gemm_bf16<0, false, true><<<gemm_grid, 256, 0, stream>>>(sg, WT2, b2, obuf, out, nullptr, M, d, d);
}
